// SelfAttention_86552180949746
// MI455X (gfx1250) — matrix operand order repaired and checked by reading, not compiled, not run on hardware
//
#include <hip/hip_runtime.h>

// Problem constants (match reference)
#define BB 4
#define SS 2048
#define EE 512
#define HH 8
#define DD 64

typedef __attribute__((ext_vector_type(16))) __bf16 v16bf;
typedef __attribute__((ext_vector_type(8)))  float  v8f;

#ifndef __has_builtin
#define __has_builtin(x) 0
#endif

#if __has_builtin(__builtin_amdgcn_global_load_async_to_lds_b128)
#define HAVE_ASYNC_COPY 1
#else
#define HAVE_ASYNC_COPY 0
#endif

#if __has_builtin(__builtin_amdgcn_s_wait_asynccnt)
#define WAIT_ASYNC(n) __builtin_amdgcn_s_wait_asynccnt(n)
#else
#define WAIT_ASYNC(n) asm volatile("s_wait_asynccnt %0" ::"n"(n) : "memory")
#endif

// 16-byte global -> LDS copy (async on CDNA5, tracked by ASYNCcnt).
// Builtin payload type (from compiler diagnostic): <4 x i32>, src in AS1, dst in AS3.
#if HAVE_ASYNC_COPY
typedef int av4i __attribute__((vector_size(16)));
#endif
static __device__ inline void async_copy16(const __bf16* g, __bf16* l) {
#if HAVE_ASYNC_COPY
  __builtin_amdgcn_global_load_async_to_lds_b128(
      (__attribute__((address_space(1))) av4i*)(g),
      (__attribute__((address_space(3))) av4i*)(l), 0, 0);
#else
  *(uint4*)l = *(const uint4*)g;
#endif
}

// ---------------------------------------------------------------------------
// WMMA helper: D = A(16x32 bf16) x B(32x16 bf16) + C(16x16 f32)
// ---------------------------------------------------------------------------
static __device__ inline v8f wmma_bf16(v16bf a, v16bf b, v8f c) {
  return __builtin_amdgcn_wmma_f32_16x16x32_bf16(
      /*neg_a=*/false, a, /*neg_b=*/false, b,
      /*c_mod=*/(short)0, c, /*reuse_a=*/false, /*reuse_b=*/false);
}

// ---------------------------------------------------------------------------
// Fragment loaders (wave32; A/B use this code's contiguous k order, same for both)
//
// A 16x32 bf16: lane -> M = lane&15, half = lane>>4.
//   element j in [0,8)  : K = half*16 + j
//   element j in [8,16) : K = half*16 + 8 + (j-8)
// B 32x16 bf16: lane -> N = lane&15, khalf = lane>>4.
//   element j : K = khalf*16 + j
// C/D 16x16 f32: VGPR r, lanes 0-15: (M=r, N=lane); lanes 16-31: (M=8+r, N=lane-16)
// ---------------------------------------------------------------------------
static __device__ inline v16bf load_a_f32(const float* __restrict__ base, int ld,
                                          int m0, int k0, int lane) {
  const float* r = base + (m0 + (lane & 15)) * ld + k0 + ((lane >> 4) << 4);
  v16bf a;
#pragma unroll
  for (int j = 0; j < 8; ++j) a[j] = (__bf16)r[j];
#pragma unroll
  for (int j = 0; j < 8; ++j) a[8 + j] = (__bf16)r[8 + j];
  return a;
}

static __device__ inline v16bf load_a_bf16(const __bf16* base, int ld,
                                           int m0, int k0, int lane) {
  const __bf16* r = base + (m0 + (lane & 15)) * ld + k0 + ((lane >> 4) << 4);
  v16bf a;
#pragma unroll
  for (int j = 0; j < 8; ++j) a[j] = r[j];
#pragma unroll
  for (int j = 0; j < 8; ++j) a[8 + j] = r[8 + j];
  return a;
}

// B[k][n] = base[(n0+n)*ld + k0 + k]  (columns of B are rows of `base`)
static __device__ inline v16bf load_b_f32rows(const float* __restrict__ base, int ld,
                                              int n0, int k0, int lane) {
  const float* r = base + (n0 + (lane & 15)) * ld + k0 + ((lane >> 4) << 4);
  v16bf b;
#pragma unroll
  for (int j = 0; j < 16; ++j) b[j] = (__bf16)r[j];
  return b;
}

static __device__ inline v16bf load_b_bf16rows(const __bf16* base, int ld,
                                               int n0, int k0, int lane) {
  const __bf16* r = base + (n0 + (lane & 15)) * ld + k0 + ((lane >> 4) << 4);
  v16bf b;
#pragma unroll
  for (int j = 0; j < 16; ++j) b[j] = r[j];
  return b;
}

// ---------------------------------------------------------------------------
// Kernel 1: input projections.  Y = X @ W^T + bias   (M=B*S, N=E, K=E)
// mode 0: store Y as bf16 [B,H,S,D]   (for Q, K)
// mode 1: store Y as bf16 [B,H,D,S]   (transposed V, so A.V B-frags are contiguous)
// ---------------------------------------------------------------------------
__global__ __launch_bounds__(256) void proj_kernel(const float* __restrict__ X,
                                                   const float* __restrict__ W,
                                                   const float* __restrict__ bias,
                                                   __bf16* __restrict__ Y, int mode) {
  const int wid  = (blockIdx.x * blockDim.x + threadIdx.x) >> 5;
  const int lane = threadIdx.x & 31;
  const int tm = wid >> 5;       // EE/16 == 32 tiles in N
  const int tn = wid & 31;
  const int m0 = tm * 16, n0 = tn * 16;

  v8f c = {};
#pragma unroll
  for (int k0 = 0; k0 < EE; k0 += 32) {
    v16bf a = load_a_f32(X, EE, m0, k0, lane);
    v16bf b = load_b_f32rows(W, EE, n0, k0, lane);  // B[k][n] = W[n][k]
    c = wmma_bf16(a, b, c);
  }

  const int n  = n0 + (lane & 15);
  const int hi = lane >> 4;
  const float bv = bias[n];
  const int h = n >> 6, d = n & (DD - 1);
#pragma unroll
  for (int r = 0; r < 8; ++r) {
    const int m = m0 + r + (hi << 3);
    const int bidx = m >> 11;            // / SS
    const int s    = m & (SS - 1);
    const float val = c[r] + bv;
    if (mode == 0)
      Y[(((size_t)(bidx * HH + h)) * SS + s) * DD + d] = (__bf16)val;
    else
      Y[(((size_t)(bidx * HH + h)) * DD + d) * SS + s] = (__bf16)val;
  }
}

// ---------------------------------------------------------------------------
// Kernel 2: flash-style attention per head, block-cooperative K/V staging.
//
// Each block = 8 waves = 8 consecutive 16-row query tiles of ONE (b,h).
// K/V are streamed in 32-column chunks, DOUBLE-BUFFERED in LDS via
// GLOBAL_LOAD_ASYNC_TO_LDS_B128 (ASYNCcnt): issue chunk i+1, wait asynccnt<=2
// (async loads complete in order -> chunk i landed), barrier, compute chunk i.
// Scores never touch HBM; online softmax via shfl_xor inside 16-lane halves.
// ---------------------------------------------------------------------------
struct AttnLds {
  __bf16 K[2][32 * DD];   // 2 x 4KB: chunk rows kc..kc+31, [row][d]
  __bf16 V[2][DD * 32];   // 2 x 4KB: [d][s-rel]
  __bf16 P[8][16 * 32];   // per-wave private P-relayout tiles
};

static __device__ inline void issue_chunk(const __bf16* Kb, const __bf16* Vb, int kc,
                                          __bf16* Kl, __bf16* Vl, int tid) {
  {  // K chunk: 32 rows x 64 bf16 = 256 x 16B
    const int row = tid >> 3, seg = (tid & 7) << 3;
    async_copy16(Kb + (size_t)(kc + row) * DD + seg, Kl + row * DD + seg);
  }
  {  // V chunk: 64 rows(d) x 32 bf16 = 256 x 16B
    const int d = tid >> 2, seg = (tid & 3) << 3;
    async_copy16(Vb + (size_t)d * SS + kc + seg, Vl + d * 32 + seg);
  }
}

__global__ __launch_bounds__(256) void attn_kernel(const __bf16* __restrict__ Qh,
                                                   const __bf16* __restrict__ Kh,
                                                   const __bf16* __restrict__ Vt,
                                                   const int* __restrict__ mask,
                                                   __bf16* __restrict__ Ch) {
  __shared__ AttnLds lds;

  const int tid   = threadIdx.x;
  const int wslot = tid >> 5;
  const int lane  = tid & 31;

  const int bh = blockIdx.x >> 4;                    // 16 blocks per (b,h)
  const int b  = bh >> 3;
  const int h  = bh & 7;
  const int q0 = (((blockIdx.x & 15) << 3) + wslot) * 16;

  const __bf16* Qb = Qh + (size_t)bh * SS * DD;
  const __bf16* Kb = Kh + (size_t)bh * SS * DD;
  const __bf16* Vb = Vt + (size_t)bh * DD * SS;      // [D, S] per head

  // Q A-fragments for the whole tile (16x64 = two K=32 fragments), loaded once.
  const v16bf aQ0 = load_a_bf16(Qb, DD, q0, 0, lane);
  const v16bf aQ1 = load_a_bf16(Qb, DD, q0, 32, lane);

  v8f acc0 = {}, acc1 = {}, acc2 = {}, acc3 = {};    // O tile 16x64 f32
  float mrow[8], lrow[8];
#pragma unroll
  for (int r = 0; r < 8; ++r) { mrow[r] = -3.0e38f; lrow[r] = 0.0f; }

  const float SCALE  = 0.044194173824159216f;          // 1/sqrt(512)
  const float NEGBIG = -1e19f * 0.044194173824159216f; // masked_fill then /sqrt(E)
  const float L2E    = 1.44269504088896341f;
  const int hi = lane >> 4;
  const int nn = lane & 15;

  // prologue: stage chunk 0
  issue_chunk(Kb, Vb, 0, lds.K[0], lds.V[0], tid);

  for (int kc = 0; kc < SS; kc += 32) {
    const int cur  = (kc >> 5) & 1;
    const bool more = (kc + 32) < SS;
    if (more) issue_chunk(Kb, Vb, kc + 32, lds.K[cur ^ 1], lds.V[cur ^ 1], tid);
#if HAVE_ASYNC_COPY
    if (more) { WAIT_ASYNC(2); } else { WAIT_ASYNC(0); }
#endif
    __syncthreads();   // chunk `cur` visible to all waves

    const __bf16* Kl = lds.K[cur];
    const __bf16* Vl = lds.V[cur];

    // ---- scores: S(16x32) = Q(16x64) . K^T(64x32), 4 WMMAs (B-frags from LDS)
    v8f s0 = {}, s1 = {};
    {
      v16bf bk = load_b_bf16rows(Kl, DD, 0, 0, lane);   // B[d][n] = Kl[n][d]
      s0 = wmma_bf16(aQ0, bk, s0);
      bk = load_b_bf16rows(Kl, DD, 0, 32, lane);
      s0 = wmma_bf16(aQ1, bk, s0);
      bk = load_b_bf16rows(Kl, DD, 16, 0, lane);
      s1 = wmma_bf16(aQ0, bk, s1);
      bk = load_b_bf16rows(Kl, DD, 16, 32, lane);
      s1 = wmma_bf16(aQ1, bk, s1);
    }

    // ---- mask (before scaling, faithful to reference) + scale
    float p0[8], p1[8], fac[8];
#pragma unroll
    for (int r = 0; r < 8; ++r) {
      const int q = q0 + r + (hi << 3);
      const int* mr = mask + ((size_t)b * SS + q) * SS + kc;
      p0[r] = mr[nn]      ? s0[r] * SCALE : NEGBIG;
      p1[r] = mr[16 + nn] ? s1[r] * SCALE : NEGBIG;
      if (r == 0 && kc + 32 < SS)
        __builtin_prefetch(mr + 32 + nn, 0, 1);   // global_prefetch for next mask chunk
    }

    // ---- online softmax update (row reductions inside 16-lane half-groups)
#pragma unroll
    for (int r = 0; r < 8; ++r) {
      float cm = fmaxf(p0[r], p1[r]);
      cm = fmaxf(cm, __shfl_xor(cm, 8, 32));
      cm = fmaxf(cm, __shfl_xor(cm, 4, 32));
      cm = fmaxf(cm, __shfl_xor(cm, 2, 32));
      cm = fmaxf(cm, __shfl_xor(cm, 1, 32));
      const float mnew = fmaxf(mrow[r], cm);
      fac[r]  = exp2f((mrow[r] - mnew) * L2E);
      mrow[r] = mnew;
      p0[r] = exp2f((p0[r] - mnew) * L2E);
      p1[r] = exp2f((p1[r] - mnew) * L2E);
      float rs = p0[r] + p1[r];
      rs += __shfl_xor(rs, 8, 32);
      rs += __shfl_xor(rs, 4, 32);
      rs += __shfl_xor(rs, 2, 32);
      rs += __shfl_xor(rs, 1, 32);
      lrow[r] = lrow[r] * fac[r] + rs;
    }
#pragma unroll
    for (int r = 0; r < 8; ++r) {
      acc0[r] *= fac[r]; acc1[r] *= fac[r]; acc2[r] *= fac[r]; acc3[r] *= fac[r];
    }

    // ---- relayout P (C layout -> A layout) through per-wave LDS tile, as bf16
    __bf16* my = &lds.P[wslot][0];
#pragma unroll
    for (int r = 0; r < 8; ++r) {
      const int m = r + (hi << 3);
      my[m * 32 + nn]      = (__bf16)p0[r];
      my[m * 32 + 16 + nn] = (__bf16)p1[r];
    }
    asm volatile("" ::: "memory");   // LDS ops are in-order within a wave; stop compiler reordering
    const v16bf aP = load_a_bf16(my, 32, 0, 0, lane);
    asm volatile("" ::: "memory");

    // ---- O += P(16x32) . V(32x64), 4 WMMAs (B-frags from LDS, Vl is [d][s-rel])
    v16bf bv = load_b_bf16rows(Vl, 32, 0, 0, lane);
    acc0 = wmma_bf16(aP, bv, acc0);
    bv = load_b_bf16rows(Vl, 32, 16, 0, lane);
    acc1 = wmma_bf16(aP, bv, acc1);
    bv = load_b_bf16rows(Vl, 32, 32, 0, lane);
    acc2 = wmma_bf16(aP, bv, acc2);
    bv = load_b_bf16rows(Vl, 32, 48, 0, lane);
    acc3 = wmma_bf16(aP, bv, acc3);

    __syncthreads();   // everyone done reading chunk `cur` before it is refilled
  }

  // ---- finalize: divide by row sums, write context bf16 [B,S,E] (head-concat)
#pragma unroll
  for (int r = 0; r < 8; ++r) {
    const int q = q0 + r + (hi << 3);
    const float inv = 1.0f / lrow[r];
    __bf16* orow = Ch + ((size_t)b * SS + q) * EE + h * DD;
    orow[nn]      = (__bf16)(acc0[r] * inv);
    orow[16 + nn] = (__bf16)(acc1[r] * inv);
    orow[32 + nn] = (__bf16)(acc2[r] * inv);
    orow[48 + nn] = (__bf16)(acc3[r] * inv);
  }
}

// ---------------------------------------------------------------------------
// Kernel 3: output projection.  out = Ch @ Wo^T + bo  -> fp32 d_out
// ---------------------------------------------------------------------------
__global__ __launch_bounds__(256) void outproj_kernel(const __bf16* X,
                                                      const float* __restrict__ W,
                                                      const float* __restrict__ bias,
                                                      float* __restrict__ Out) {
  const int wid  = (blockIdx.x * blockDim.x + threadIdx.x) >> 5;
  const int lane = threadIdx.x & 31;
  const int tm = wid >> 5;
  const int tn = wid & 31;
  const int m0 = tm * 16, n0 = tn * 16;

  v8f c = {};
#pragma unroll
  for (int k0 = 0; k0 < EE; k0 += 32) {
    v16bf a = load_a_bf16(X, EE, m0, k0, lane);
    v16bf b = load_b_f32rows(W, EE, n0, k0, lane);
    c = wmma_bf16(a, b, c);
  }

  const int n  = n0 + (lane & 15);
  const int hi = lane >> 4;
  const float bv = bias[n];
#pragma unroll
  for (int r = 0; r < 8; ++r) {
    const int m = m0 + r + (hi << 3);
    Out[(size_t)m * EE + n] = c[r] + bv;
  }
}

// ---------------------------------------------------------------------------
// Host launcher
// ---------------------------------------------------------------------------
extern "C" void kernel_launch(void* const* d_in, const int* in_sizes, int n_in,
                              void* d_out, int out_size, void* d_ws, size_t ws_size,
                              hipStream_t stream) {
  (void)in_sizes; (void)n_in; (void)out_size; (void)ws_size;
  const float* values  = (const float*)d_in[0];
  const float* keys    = (const float*)d_in[1];
  const float* queries = (const float*)d_in[2];
  const int*   mask    = (const int*)d_in[3];
  const float* Wv = (const float*)d_in[4];
  const float* bv = (const float*)d_in[5];
  const float* Wk = (const float*)d_in[6];
  const float* bk = (const float*)d_in[7];
  const float* Wq = (const float*)d_in[8];
  const float* bq = (const float*)d_in[9];
  const float* Wo = (const float*)d_in[10];
  const float* bo = (const float*)d_in[11];
  float* out = (float*)d_out;

  // Workspace layout (bf16): Qh[B,H,S,D] | Kh[B,H,S,D] | Vt[B,H,D,S] | Ch[B,S,E]
  char* ws = (char*)d_ws;
  const size_t seg = (size_t)BB * HH * SS * DD * sizeof(__bf16);  // 8 MB each
  __bf16* Qh = (__bf16*)(ws + 0 * seg);
  __bf16* Kh = (__bf16*)(ws + 1 * seg);
  __bf16* Vt = (__bf16*)(ws + 2 * seg);
  __bf16* Ch = (__bf16*)(ws + 3 * seg);

  const dim3 blk(256);
  // (B*S/16)*(E/16) = 16384 tiles / 8 waves per block = 2048 blocks
  proj_kernel<<<2048, blk, 0, stream>>>(queries, Wq, bq, Qh, 0);
  proj_kernel<<<2048, blk, 0, stream>>>(keys,    Wk, bk, Kh, 0);
  proj_kernel<<<2048, blk, 0, stream>>>(values,  Wv, bv, Vt, 1);
  // B*H*(S/16) = 4096 waves / 8 per block = 512 blocks (exact, no tail)
  attn_kernel<<<512, blk, 0, stream>>>(Qh, Kh, Vt, mask, Ch);
  outproj_kernel<<<2048, blk, 0, stream>>>(Ch, Wo, bo, out);
}
